// VectorQuantizer_7275674600257
// MI455X (gfx1250) — matrix operand order repaired and checked by reading, not compiled, not run on hardware
//
#include <hip/hip_runtime.h>
#include <hip/hip_bf16.h>
#include <math.h>

// Problem constants (from reference)
#define B_ROWS        16384
#define NUM_CODES     4096
#define EMBED         512
#define NORMAL_COUNT  2048   // NUM_CODES/2

typedef __attribute__((ext_vector_type(16))) __bf16 v16bf;
typedef __attribute__((ext_vector_type(8)))  float  v8f;

// round-to-nearest-even f32 -> bf16
__device__ __forceinline__ unsigned short f2bf(float f) {
  union { float f; unsigned u; } v; v.f = f;
  unsigned r = v.u + 0x7FFFu + ((v.u >> 16) & 1u);
  return (unsigned short)(r >> 16);
}

// ---------------------------------------------------------------------------
// Prepass: per-row f32 -> bf16 conversion + squared-norm (one wave per row).
// ---------------------------------------------------------------------------
__global__ void vq_cvt_norm_kernel(const float* __restrict__ src,
                                   unsigned short* __restrict__ dst,
                                   float* __restrict__ norms, int nrows) {
  const int wave = (int)((blockIdx.x * blockDim.x + threadIdx.x) >> 5);
  const int lane = (int)(threadIdx.x & 31);
  if (wave >= nrows) return;
  const float2* row = (const float2*)(src + (size_t)wave * EMBED);
  unsigned int* out = (unsigned int*)(dst + (size_t)wave * EMBED);
  float sq = 0.f;
#pragma unroll
  for (int j = 0; j < 8; ++j) {                 // 32 lanes * 8 float2 = 512
    float2 p = row[lane + 32 * j];
    sq += p.x * p.x + p.y * p.y;
    out[lane + 32 * j] = (unsigned)f2bf(p.x) | ((unsigned)f2bf(p.y) << 16);
  }
#pragma unroll
  for (int m = 16; m >= 1; m >>= 1) sq += __shfl_xor(sq, m, 32);
  if (lane == 0) norms[wave] = sq;
}

// Zero the three scalar-loss accumulator slots.
__global__ void vq_init_kernel(float* __restrict__ out) {
  if (threadIdx.x < 3)
    out[(size_t)B_ROWS * EMBED + B_ROWS + threadIdx.x] = 0.f;
}

// ---------------------------------------------------------------------------
// Fused main kernel. Per wave: a 32-row z_e block (NR=2 row tiles) swept
// against all codes, NC=2 column tiles per iteration:
//   per k-chunk: 4 A-loads + 4 B-loads feed 4 WMMAs (1KB operands/WMMA)
//   and 4 independent accumulator chains keep the XDL pipeline busy.
// Column sweep split into normal/abnormal phases (no per-element masking);
// per-phase cross-lane argmin reduction commits only the winning index.
// ---------------------------------------------------------------------------
__global__ void __launch_bounds__(128, 1)
vq_main_kernel(const float* __restrict__ z_e,
               const int* __restrict__ labels,
               const float* __restrict__ codebook,
               const unsigned short* __restrict__ zbf,
               const unsigned short* __restrict__ cbf,
               const float* __restrict__ znorm,
               const float* __restrict__ cnorm,
               float* __restrict__ out) {
  const int lane  = (int)(threadIdx.x & 31);
  const int wave  = (int)(threadIdx.x >> 5);
  const int row0  = (int)blockIdx.x * 128 + wave * 32;  // 4 waves * 32 rows
  const int half  = lane >> 4;        // 0: lanes 0-15, 1: lanes 16-31
  const int l15   = lane & 15;
  const int rbase = half * 8;         // C-matrix: VGPR r holds row r + 8*half

  // Persistent per-row state: 2 row tiles x 8 rows per half-wave lane.
  float zn[2][8], sum_n[2][8], sum_a[2][8];
  int   res_i[2][8];
  int   labBits = 0;                  // bit (t*8+r) => row is a normal sample
#pragma unroll
  for (int t = 0; t < 2; ++t) {
#pragma unroll
    for (int r = 0; r < 8; ++r) {
      const int rg = row0 + t * 16 + rbase + r;
      zn[t][r] = znorm[rg];
      if (labels[rg] == 0) labBits |= 1 << (t * 8 + r);
      sum_n[t][r] = 0.f; sum_a[t][r] = 0.f;
      res_i[t][r] = 0;
    }
  }

  // A (z_e) per-lane bases, 16-bit A 16x32 layout (ISA 7.12.2):
  //   lanes 0-15  -> row l15, K = {k0..k0+7, k0+16..k0+23}
  //   lanes 16-31 -> row l15, K = {k0+8..k0+15, k0+24..k0+31}
  const unsigned short* aptr0 = zbf + (size_t)(row0 + l15) * EMBED + half * 8;
  const unsigned short* aptr1 = aptr0 + (size_t)16 * EMBED;  // rows +16

  union U16 { v16bf v; uint4 q[2]; };

#pragma unroll
  for (int phase = 0; phase < 2; ++phase) {
    // Phase-local unconditional argmin state.
    float tmp_d[2][8];
    int   tmp_i[2][8];
#pragma unroll
    for (int t = 0; t < 2; ++t)
#pragma unroll
      for (int r = 0; r < 8; ++r) { tmp_d[t][r] = 3.0e38f; tmp_i[t][r] = 0x7fffffff; }

    // exp-sums go straight into the class bucket for this phase (no temps).
    float (*sumP)[8] = (phase == 0) ? sum_n : sum_a;

    const int base = phase * NORMAL_COUNT;
    for (int col0 = base; col0 < base + NORMAL_COUNT; col0 += 32) {
      const float cn0 = cnorm[col0 + l15];
      const float cn1 = cnorm[col0 + 16 + l15];
      // B (codebook^T): lanes 0-15 hold K={k0..k0+7, k0+16..k0+23} of code col,
      // lanes 16-31 hold K={k0+8..k0+15, k0+24..k0+31} (matches A), no transpose.
      const unsigned short* bp0 = cbf + (size_t)(col0 + l15) * EMBED + half * 8;
      const unsigned short* bp1 = bp0 + (size_t)16 * EMBED;

      v8f acc00 = {}, acc01 = {}, acc10 = {}, acc11 = {};
#pragma unroll 2
      for (int k0 = 0; k0 < EMBED; k0 += 32) {
        U16 A0, A1, B0, B1;
        A0.q[0] = *(const uint4*)(aptr0 + k0);
        A0.q[1] = *(const uint4*)(aptr0 + k0 + 16);
        A1.q[0] = *(const uint4*)(aptr1 + k0);
        A1.q[1] = *(const uint4*)(aptr1 + k0 + 16);
        B0.q[0] = *(const uint4*)(bp0 + k0);
        B0.q[1] = *(const uint4*)(bp0 + k0 + 16);
        B1.q[0] = *(const uint4*)(bp1 + k0);
        B1.q[1] = *(const uint4*)(bp1 + k0 + 16);
        acc00 = __builtin_amdgcn_wmma_f32_16x16x32_bf16(
            false, A0.v, false, B0.v, (short)0, acc00, false, false);
        acc10 = __builtin_amdgcn_wmma_f32_16x16x32_bf16(
            false, A1.v, false, B0.v, (short)0, acc10, false, false);
        acc01 = __builtin_amdgcn_wmma_f32_16x16x32_bf16(
            false, A0.v, false, B1.v, (short)0, acc01, false, false);
        acc11 = __builtin_amdgcn_wmma_f32_16x16x32_bf16(
            false, A1.v, false, B1.v, (short)0, acc11, false, false);
      }

      // Epilogue: scores -> distances -> exp-sum + running argmin (no masks).
#pragma unroll
      for (int r = 0; r < 8; ++r) {
        const float d00 =
            __builtin_amdgcn_sqrtf(fmaxf(zn[0][r] + cn0 - 2.f * acc00[r], 0.f));
        const float d01 =
            __builtin_amdgcn_sqrtf(fmaxf(zn[0][r] + cn1 - 2.f * acc01[r], 0.f));
        const float d10 =
            __builtin_amdgcn_sqrtf(fmaxf(zn[1][r] + cn0 - 2.f * acc10[r], 0.f));
        const float d11 =
            __builtin_amdgcn_sqrtf(fmaxf(zn[1][r] + cn1 - 2.f * acc11[r], 0.f));
        sumP[0][r] += __expf(-d00) + __expf(-d01);  // d in [0,~30]: f32-safe
        sumP[1][r] += __expf(-d10) + __expf(-d11);
        if (d00 < tmp_d[0][r]) { tmp_d[0][r] = d00; tmp_i[0][r] = col0 + l15; }
        if (d01 < tmp_d[0][r]) { tmp_d[0][r] = d01; tmp_i[0][r] = col0 + 16 + l15; }
        if (d10 < tmp_d[1][r]) { tmp_d[1][r] = d10; tmp_i[1][r] = col0 + l15; }
        if (d11 < tmp_d[1][r]) { tmp_d[1][r] = d11; tmp_i[1][r] = col0 + 16 + l15; }
      }
    }

    // Per-phase cross-lane argmin reduction (masks < 16 stay inside halves),
    // then commit only the index for rows whose label matches this phase.
#pragma unroll
    for (int m = 8; m >= 1; m >>= 1) {
#pragma unroll
      for (int t = 0; t < 2; ++t)
#pragma unroll
        for (int r = 0; r < 8; ++r) {
          const float od = __shfl_xor(tmp_d[t][r], m, 32);
          const int   oi = __shfl_xor(tmp_i[t][r], m, 32);
          if (od < tmp_d[t][r] || (od == tmp_d[t][r] && oi < tmp_i[t][r])) {
            tmp_d[t][r] = od; tmp_i[t][r] = oi;
          }
        }
    }
#pragma unroll
    for (int t = 0; t < 2; ++t)
#pragma unroll
      for (int r = 0; r < 8; ++r) {
        const bool wantNormal = ((labBits >> (t * 8 + r)) & 1) != 0;
        const bool commit = (phase == 0) ? wantNormal : !wantNormal;
        if (commit) res_i[t][r] = tmp_i[t][r];
      }
  }

  // Butterfly-reduce the softmax sums across the 16 lanes of each half.
#pragma unroll
  for (int m = 8; m >= 1; m >>= 1) {
#pragma unroll
    for (int t = 0; t < 2; ++t)
#pragma unroll
      for (int r = 0; r < 8; ++r) {
        sum_n[t][r] += __shfl_xor(sum_n[t][r], m, 32);
        sum_a[t][r] += __shfl_xor(sum_a[t][r], m, 32);
      }
  }

  float* idx_out  = out + (size_t)B_ROWS * EMBED;
  float* loss_out = idx_out + B_ROWS;   // [codebook, commitment, cls]

  // One lane per half writes indices + classification loss for its 16 rows.
  if (l15 == 0) {
    float clsPart = 0.f;
#pragma unroll
    for (int t = 0; t < 2; ++t) {
#pragma unroll
      for (int r = 0; r < 8; ++r) {
        const int rg = row0 + t * 16 + rbase + r;
        idx_out[rg] = (float)res_i[t][r];
        float p = sum_n[t][r] / (sum_n[t][r] + sum_a[t][r]);
        p = fminf(fmaxf(p, 1e-12f), 1.f - 1e-12f);
        const float tl = ((labBits >> (t * 8 + r)) & 1) ? 1.f : 0.f;
        clsPart += -(tl * __logf(p) + (1.f - tl) * __logf(1.f - p));
      }
    }
    atomicAdd(loss_out + 2, clsPart * (1.f / (float)B_ROWS));
  }

  // Gather z_q = codebook[argmin] (f32), write z_q_st (== z_q numerically),
  // and accumulate (z_e - z_q)^2 for both quantization losses (equal values).
  float lsq = 0.f;
#pragma unroll
  for (int row = 0; row < 32; ++row) {
    const int t  = row >> 4;
    const int rr = row & 15;
    const int idx = (rr < 8) ? __shfl(res_i[t][rr], 0, 32)
                             : __shfl(res_i[t][rr - 8], 16, 32);
    const int rg = row0 + row;
    const float4* cbrow = (const float4*)(codebook + (size_t)idx * EMBED);
    const float4* zrow  = (const float4*)(z_e + (size_t)rg * EMBED);
    float4*       orow  = (float4*)(out + (size_t)rg * EMBED);
#pragma unroll
    for (int j = 0; j < 4; ++j) {                // 32 lanes * 4 float4 = 512
      const float4 q = cbrow[lane + 32 * j];
      const float4 z = zrow[lane + 32 * j];
      orow[lane + 32 * j] = q;
      const float dx = z.x - q.x, dy = z.y - q.y;
      const float dz = z.z - q.z, dw = z.w - q.w;
      lsq += dx * dx + dy * dy + dz * dz + dw * dw;
    }
  }
#pragma unroll
  for (int m = 16; m >= 1; m >>= 1) lsq += __shfl_xor(lsq, m, 32);
  if (lane == 0) {
    const float v = lsq * (1.f / ((float)B_ROWS * (float)EMBED));
    atomicAdd(loss_out + 0, v);
    atomicAdd(loss_out + 1, v);
  }
}

// ---------------------------------------------------------------------------
extern "C" void kernel_launch(void* const* d_in, const int* in_sizes, int n_in,
                              void* d_out, int out_size, void* d_ws, size_t ws_size,
                              hipStream_t stream) {
  const float* z_e      = (const float*)d_in[0];  // (B, D) f32
  const int*   labels   = (const int*)d_in[1];    // (B,)   i32
  const float* codebook = (const float*)d_in[2];  // (K, D) f32
  float*       out      = (float*)d_out;          // [z_q_st | indices | 3 losses]

  // Workspace layout: bf16 z_e | bf16 codebook | z norms | code norms (~21 MB)
  unsigned short* zbf   = (unsigned short*)d_ws;
  unsigned short* cbf   = zbf + (size_t)B_ROWS * EMBED;
  float*          znorm = (float*)(cbf + (size_t)NUM_CODES * EMBED);
  float*          cnorm = znorm + B_ROWS;

  // 1) bf16 conversion + row norms (one wave per row, 8 waves per block)
  vq_cvt_norm_kernel<<<B_ROWS / 8, 256, 0, stream>>>(z_e, zbf, znorm, B_ROWS);
  vq_cvt_norm_kernel<<<NUM_CODES / 8, 256, 0, stream>>>(codebook, cbf, cnorm,
                                                        NUM_CODES);
  // 2) zero the loss accumulators
  vq_init_kernel<<<1, 32, 0, stream>>>(out);
  // 3) fused WMMA distance + softmax + argmin + gather (32 rows per wave)
  vq_main_kernel<<<B_ROWS / 128, 128, 0, stream>>>(z_e, labels, codebook, zbf,
                                                   cbf, znorm, cnorm, out);
}
